// GATImputer_17712445129456
// MI455X (gfx1250) — matrix operand order repaired and checked by reading, not compiled, not run on hardware
//
#include <hip/hip_runtime.h>
#include <hip/hip_bf16.h>
#include <stdint.h>

typedef _Float16 half_t;
typedef __attribute__((ext_vector_type(16))) _Float16 v16h;
typedef __attribute__((ext_vector_type(8)))  float    v8f;

constexpr int Bn   = 8;
constexpr int Dn   = 1024;   // nodes
constexpr int Tn   = 512;
constexpr int HIDn = 256;
constexpr int NHn  = 4;
constexpr int OUTn = 64;
constexpr float ALPHAc = 0.2f;
constexpr float EPSc   = 1e-5f;

__device__ __forceinline__ float lrelu(float x) { return x > 0.0f ? x : ALPHAc * x; }

// ---------------------------------------------------------------------------
// 1) Imputation: pos_mean over batch, fill missing, concat [X_mean, mask] -> f16
//    x16 layout: (B*Dn, 2*Tn) row-major (K-major for the GEMM A operand)
// ---------------------------------------------------------------------------
__global__ void k_impute_concat(const float* __restrict__ X,
                                const float* __restrict__ Mk,
                                half_t* __restrict__ x16)
{
    int idx = blockIdx.x * blockDim.x + threadIdx.x;   // over Dn*Tn
    if (idx >= Dn * Tn) return;
    int d = idx / Tn, t = idx % Tn;
    float xs[Bn], ms[Bn];
    float sum = 0.f, cnt = 0.f;
#pragma unroll
    for (int b = 0; b < Bn; ++b) {
        size_t o = ((size_t)b * Dn + d) * Tn + t;
        float xv = X[o], mv = Mk[o];
        xs[b] = xv; ms[b] = mv;
        sum += xv * mv; cnt += mv;
    }
    float mean = sum / (cnt + 1e-10f);
#pragma unroll
    for (int b = 0; b < Bn; ++b) {
        float xm = xs[b] * ms[b] + (1.0f - ms[b]) * mean;
        size_t row = (size_t)b * Dn + d;
        x16[row * (2 * Tn) + t]      = (half_t)xm;
        x16[row * (2 * Tn) + Tn + t] = (half_t)ms[b];
    }
}

// ---------------------------------------------------------------------------
// 2) Transpose + convert: Bt[n,k] = W[k,n]  (f32 (K,N) -> f16 (N,K))
// ---------------------------------------------------------------------------
__global__ void k_transpose_cvt(const float* __restrict__ W, half_t* __restrict__ Bt,
                                int K, int N)
{
    int idx = blockIdx.x * blockDim.x + threadIdx.x;   // over N*K
    if (idx >= N * K) return;
    int n = idx / K, k = idx % K;
    Bt[idx] = (half_t)W[(size_t)k * N + n];
}

// Pack gat_W[l] (NH,HID,OUT) -> WcatT (N'=NH*OUT, K=HID) f16, N' index = h*OUT+e
__global__ void k_pack_gatW(const float* __restrict__ gW, half_t* __restrict__ WcatT)
{
    int idx = blockIdx.x * blockDim.x + threadIdx.x;   // over HID*HID
    if (idx >= HIDn * HIDn) return;
    int n = idx / HIDn, k = idx % HIDn;
    int hh = n / OUTn, e = n % OUTn;
    WcatT[idx] = (half_t)gW[((size_t)hh * HIDn + k) * OUTn + e];
}

// Plain elementwise f32 -> f16
__global__ void k_cvt_f16(const float* __restrict__ in, half_t* __restrict__ out, int n)
{
    int idx = blockIdx.x * blockDim.x + threadIdx.x;
    if (idx < n) out[idx] = (half_t)in[idx];
}

// hW (B*Dn, HID) f32 -> hWt (B*NH, OUT, Dn) f16  (transposed V for attention GEMM)
__global__ void k_cvt_hWt(const float* __restrict__ hW, half_t* __restrict__ hWt)
{
    int idx = blockIdx.x * blockDim.x + threadIdx.x;   // over B*NH*OUT*Dn
    if (idx >= Bn * NHn * OUTn * Dn) return;
    int m  = idx % Dn;
    int e  = (idx / Dn) % OUTn;
    int hh = (idx / (Dn * OUTn)) % NHn;
    int b  = idx / (Dn * OUTn * NHn);
    hWt[idx] = (half_t)hW[((size_t)b * Dn + m) * HIDn + hh * OUTn + e];
}

// ---------------------------------------------------------------------------
// 3) Generic WMMA GEMM: C(M,N) f32 = A(M,K) f16  x  Bt(N,K) f16  (+bias[N])
//    block = 4 waves; wave -> 16x64 strip; K step 32 via v_wmma_f32_16x16x32_f16
// ---------------------------------------------------------------------------
__global__ void __launch_bounds__(128)
k_gemm_wmma(const half_t* __restrict__ A, const half_t* __restrict__ Bt,
            const float* __restrict__ bias, float* __restrict__ C,
            int M, int N, int K)
{
    const int lane   = threadIdx.x & 31;
    const int wave   = threadIdx.x >> 5;
    const int hsel   = lane >> 4;        // half-select (lanes 0-15 vs 16-31)
    const int lane16 = lane & 15;
    const int row0   = blockIdx.x * 64 + wave * 16;
    const int col0   = blockIdx.y * 64;
    const int m      = row0 + lane16;    // A row supplied by this lane

    v8f acc[4] = {};
    const half_t* Arow = A + (size_t)m * K;

    for (int k0 = 0; k0 < K; k0 += 32) {
        // A fragment: contiguous K pairs (VGPR v, half-sel s -> K = 16*s + 2*v)
        union { v16h h; uint32_t u[8]; } af;
#pragma unroll
        for (int v = 0; v < 8; ++v) {
            int koff = k0 + 16 * hsel + 2 * v;
            af.u[v] = *(const uint32_t*)(Arow + koff);
        }
#pragma unroll
        for (int nt = 0; nt < 4; ++nt) {
            // B fragment: lane = N col, contiguous K pairs (Bt is (N,K))
            const half_t* Brow = Bt + (size_t)(col0 + nt * 16 + lane16) * K + k0 + 16 * hsel;
            v16h bf = *(const v16h*)Brow;
            acc[nt] = __builtin_amdgcn_wmma_f32_16x16x32_f16(
                false, af.h, false, bf, (short)0, acc[nt], false, false);
        }
    }
#pragma unroll
    for (int nt = 0; nt < 4; ++nt) {
        int ncol = col0 + nt * 16 + lane16;
        float bv = bias ? bias[ncol] : 0.0f;
#pragma unroll
        for (int r = 0; r < 8; ++r) {
            int mrow = row0 + r + 8 * hsel;
            C[(size_t)mrow * N + ncol] = acc[nt][r] + bv;
        }
    }
}

// ---------------------------------------------------------------------------
// 4) Attention scores: s_l/s_r dots of hW head-slices with a_left/a_right
// ---------------------------------------------------------------------------
__global__ void k_scores(const float* __restrict__ hW, const float* __restrict__ gat_a_l,
                         float* __restrict__ s_l, float* __restrict__ s_r)
{
    int idx = blockIdx.x * blockDim.x + threadIdx.x;   // (b,h,n) with n fastest
    if (idx >= Bn * NHn * Dn) return;
    int n  = idx % Dn;
    int hh = (idx / Dn) % NHn;
    int b  = idx / (Dn * NHn);
    const float* row = hW + ((size_t)b * Dn + n) * HIDn + hh * OUTn;
    const float* aL  = gat_a_l + hh * 2 * OUTn;
    const float* aR  = aL + OUTn;
    float l = 0.f, r = 0.f;
#pragma unroll 8
    for (int e = 0; e < OUTn; ++e) { l += row[e] * aL[e]; r += row[e] * aR[e]; }
    s_l[idx] = l; s_r[idx] = r;
}

// ---------------------------------------------------------------------------
// 5) Softmax stats per (b,h): rowmax via lrelu monotonicity, denom sum
// ---------------------------------------------------------------------------
__global__ void __launch_bounds__(256)
k_softmax_stats(const float* __restrict__ s_l, const float* __restrict__ s_r,
                float* __restrict__ rowmax, float* __restrict__ inv_den)
{
    __shared__ float sr[Dn];
    __shared__ float red[256];
    const int bh = blockIdx.x;
    const float* srg = s_r + (size_t)bh * Dn;
    float mx = -1e30f;
    for (int i = threadIdx.x; i < Dn; i += 256) { float v = srg[i]; sr[i] = v; mx = fmaxf(mx, v); }
    red[threadIdx.x] = mx;
    __syncthreads();
    for (int s = 128; s > 0; s >>= 1) {
        if (threadIdx.x < s) red[threadIdx.x] = fmaxf(red[threadIdx.x], red[threadIdx.x + s]);
        __syncthreads();
    }
    mx = red[0];
    for (int n = threadIdx.x; n < Dn; n += 256) {
        float sl = s_l[(size_t)bh * Dn + n];
        float rm = lrelu(sl + mx);                 // true row max (lrelu monotonic)
        float den = 0.f;
        for (int mm = 0; mm < Dn; ++mm)
            den += __expf(lrelu(sl + sr[mm]) - rm);
        rowmax[(size_t)bh * Dn + n]  = rm;
        inv_den[(size_t)bh * Dn + n] = 1.0f / den;
    }
}

// ---------------------------------------------------------------------------
// 6) Fused attention GEMM: Hout(bh, n, e) = sum_m P[n,m] * hW[bh,m,e]
//    P synthesized in-register (flash-style), V pre-transposed (bh, e, m)
// ---------------------------------------------------------------------------
__global__ void __launch_bounds__(128)
k_attn_gemm(const float* __restrict__ s_l, const float* __restrict__ s_r,
            const float* __restrict__ rowmax, const float* __restrict__ inv_den,
            const half_t* __restrict__ Vt, float* __restrict__ Hout)
{
    const int lane   = threadIdx.x & 31;
    const int wave   = threadIdx.x >> 5;
    const int hsel   = lane >> 4;
    const int lane16 = lane & 15;
    const int bh     = blockIdx.y;
    const int row0   = blockIdx.x * 64 + wave * 16;
    const int n      = row0 + lane16;

    const float  sl  = s_l[(size_t)bh * Dn + n];
    const float  rm  = rowmax[(size_t)bh * Dn + n];
    const float  rd  = inv_den[(size_t)bh * Dn + n];
    const float* srB = s_r + (size_t)bh * Dn;
    const half_t* VtB = Vt + (size_t)bh * OUTn * Dn;

    v8f acc[4] = {};
    for (int k0 = 0; k0 < Dn; k0 += 32) {
        v16h a;
#pragma unroll
        for (int v = 0; v < 8; ++v) {
            int mk = k0 + 16 * hsel + 2 * v;
            float p0 = __expf(lrelu(sl + srB[mk])     - rm) * rd;
            float p1 = __expf(lrelu(sl + srB[mk + 1]) - rm) * rd;
            a[2 * v]     = (half_t)p0;
            a[2 * v + 1] = (half_t)p1;
        }
#pragma unroll
        for (int nt = 0; nt < 4; ++nt) {
            const half_t* Brow = VtB + (size_t)(nt * 16 + lane16) * Dn + k0 + 16 * hsel;
            v16h bf = *(const v16h*)Brow;
            acc[nt] = __builtin_amdgcn_wmma_f32_16x16x32_f16(
                false, a, false, bf, (short)0, acc[nt], false, false);
        }
    }
#pragma unroll
    for (int nt = 0; nt < 4; ++nt) {
        int e = nt * 16 + lane16;
#pragma unroll
        for (int r = 0; r < 8; ++r) {
            int nr = row0 + r + 8 * hsel;
            Hout[((size_t)bh * Dn + nr) * OUTn + e] = acc[nt][r];
        }
    }
}

// ---------------------------------------------------------------------------
// 7) Residual + LayerNorm, wave-per-row (wave32 shuffles), h updated in place
// ---------------------------------------------------------------------------
__global__ void __launch_bounds__(256)
k_residual_ln(float* __restrict__ h, const float* __restrict__ hout,
              const float* __restrict__ g, const float* __restrict__ be)
{
    const int row  = blockIdx.x * 8 + (threadIdx.x >> 5);   // B*Dn rows
    const int lane = threadIdx.x & 31;
    const int b = row / Dn, n = row % Dn;

    float y[8];
    float sum = 0.f;
#pragma unroll
    for (int i = 0; i < 8; ++i) {
        int c  = lane + 32 * i;
        int hh = c >> 6, e = c & 63;
        float v = h[(size_t)row * HIDn + c] +
                  hout[((size_t)(b * NHn + hh) * Dn + n) * OUTn + e];
        y[i] = v; sum += v;
    }
#pragma unroll
    for (int off = 16; off > 0; off >>= 1) sum += __shfl_xor(sum, off, 32);
    float mu = sum * (1.0f / HIDn);
    float var = 0.f;
#pragma unroll
    for (int i = 0; i < 8; ++i) { float d = y[i] - mu; var += d * d; }
#pragma unroll
    for (int off = 16; off > 0; off >>= 1) var += __shfl_xor(var, off, 32);
    var *= (1.0f / HIDn);
    float rs = rsqrtf(var + EPSc);
#pragma unroll
    for (int i = 0; i < 8; ++i) {
        int c = lane + 32 * i;
        h[(size_t)row * HIDn + c] = (y[i] - mu) * rs * g[c] + be[c];
    }
}

// ---------------------------------------------------------------------------
extern "C" void kernel_launch(void* const* d_in, const int* in_sizes, int n_in,
                              void* d_out, int out_size, void* d_ws, size_t ws_size,
                              hipStream_t stream)
{
    const float* X_obs = (const float*)d_in[0];
    const float* mask  = (const float*)d_in[1];
    const float* W_in  = (const float*)d_in[2];
    const float* b_in  = (const float*)d_in[3];
    const float* gat_W = (const float*)d_in[4];
    const float* gat_a = (const float*)d_in[5];
    const float* ln_g  = (const float*)d_in[6];
    const float* ln_b  = (const float*)d_in[7];
    const float* W_out = (const float*)d_in[8];
    const float* b_out = (const float*)d_in[9];
    float* out = (float*)d_out;

    size_t off = 0;
    auto alloc = [&](size_t bytes) -> void* {
        void* p = (char*)d_ws + off;
        off += (bytes + 255) & ~(size_t)255;
        return p;
    };
    const int M = Bn * Dn;                              // 8192
    half_t* x16     = (half_t*)alloc((size_t)M * 2 * Tn * sizeof(half_t));
    half_t* WinT    = (half_t*)alloc((size_t)HIDn * 2 * Tn * sizeof(half_t));
    float*  h32     = (float*) alloc((size_t)M * HIDn * sizeof(float));
    half_t* h16     = (half_t*)alloc((size_t)M * HIDn * sizeof(half_t));
    half_t* WcatT   = (half_t*)alloc((size_t)HIDn * HIDn * sizeof(half_t));
    float*  hW32    = (float*) alloc((size_t)M * HIDn * sizeof(float));
    half_t* hWt16   = (half_t*)alloc((size_t)Bn * NHn * OUTn * Dn * sizeof(half_t));
    float*  s_l     = (float*) alloc((size_t)Bn * NHn * Dn * sizeof(float));
    float*  s_r     = (float*) alloc((size_t)Bn * NHn * Dn * sizeof(float));
    float*  rowmax  = (float*) alloc((size_t)Bn * NHn * Dn * sizeof(float));
    float*  inv_den = (float*) alloc((size_t)Bn * NHn * Dn * sizeof(float));
    float*  hout    = (float*) alloc((size_t)Bn * NHn * Dn * OUTn * sizeof(float));
    half_t* WoutT   = (half_t*)alloc((size_t)Tn * HIDn * sizeof(half_t));
    (void)ws_size;

    // Impute + concat -> x16 (M, 2T)
    k_impute_concat<<<(Dn * Tn + 255) / 256, 256, 0, stream>>>(X_obs, mask, x16);
    // W_in (2T,HID) -> WinT (HID,2T)
    k_transpose_cvt<<<(HIDn * 2 * Tn + 255) / 256, 256, 0, stream>>>(W_in, WinT, 2 * Tn, HIDn);
    // h = x @ W_in + b_in   (8192 x 1024 x 256)
    k_gemm_wmma<<<dim3(M / 64, HIDn / 64), 128, 0, stream>>>(x16, WinT, b_in, h32, M, HIDn, 2 * Tn);

    for (int l = 0; l < 2; ++l) {
        const float* gW_l = gat_W + (size_t)l * NHn * HIDn * OUTn;
        const float* ga_l = gat_a + (size_t)l * NHn * 2 * OUTn;
        const float* g_l  = ln_g + (size_t)l * HIDn;
        const float* b_l  = ln_b + (size_t)l * HIDn;

        k_pack_gatW<<<(HIDn * HIDn + 255) / 256, 256, 0, stream>>>(gW_l, WcatT);
        k_cvt_f16<<<(M * HIDn + 255) / 256, 256, 0, stream>>>(h32, h16, M * HIDn);
        // hW = h @ Wcat   (8192 x 256 x 256), columns grouped by head
        k_gemm_wmma<<<dim3(M / 64, HIDn / 64), 128, 0, stream>>>(h16, WcatT, nullptr, hW32, M, HIDn, HIDn);
        // V transposed per (b,h): (OUT, Dn)
        k_cvt_hWt<<<(Bn * NHn * OUTn * Dn + 255) / 256, 256, 0, stream>>>(hW32, hWt16);
        // attention scores + softmax statistics
        k_scores<<<(Bn * NHn * Dn + 255) / 256, 256, 0, stream>>>(hW32, ga_l, s_l, s_r);
        k_softmax_stats<<<Bn * NHn, 256, 0, stream>>>(s_l, s_r, rowmax, inv_den);
        // fused flash-style attention matmul (P synthesized in-register)
        k_attn_gemm<<<dim3(Dn / 64, Bn * NHn), 128, 0, stream>>>(s_l, s_r, rowmax, inv_den, hWt16, hout);
        // residual + LayerNorm
        k_residual_ln<<<M / 8, 256, 0, stream>>>(h32, hout, g_l, b_l);
    }

    // out = h @ W_out + b_out   (8192 x 256 x 512)
    k_cvt_f16<<<(M * HIDn + 255) / 256, 256, 0, stream>>>(h32, h16, M * HIDn);
    k_transpose_cvt<<<(Tn * HIDn + 255) / 256, 256, 0, stream>>>(W_out, WoutT, HIDn, Tn);
    k_gemm_wmma<<<dim3(M / 64, Tn / 64), 128, 0, stream>>>(h16, WoutT, b_out, out, M, Tn, HIDn);
}
